// RAIMT_Decoder_66984309948988
// MI455X (gfx1250) — matrix operand order repaired and checked by reading, not compiled, not run on hardware
//
#include <hip/hip_runtime.h>

typedef __bf16 bf16_t;
typedef __attribute__((ext_vector_type(16))) __bf16 v16bf;
typedef __attribute__((ext_vector_type(8)))  __bf16 v8bf;
typedef __attribute__((ext_vector_type(8)))  float  v8f;

constexpr int Dm = 512, Hh = 8, HD = 64, MEMN = 16, CHUNK = 50;
constexpr int Bb = 64, Ll = 2000, NCH = 40, T = MEMN + CHUNK; // 66

// ---------------- fragment helpers (CDNA5 wave32 WMMA; contiguous-k order used here) ----------------
// A-matrix 16x32 bf16: lane holds row (lane&15); per lane 16 contiguous elems:
//   K = lh*16 .. lh*16+15   (lh = lane>>4)
__device__ __forceinline__ v16bf afrag(const bf16_t* base /*row ptr + lh*16*/, int k) {
  v8bf lo = *(const v8bf*)(base + k);
  v8bf hi = *(const v8bf*)(base + k + 8);
  return __builtin_shufflevector(lo, hi, 0,1,2,3,4,5,6,7,8,9,10,11,12,13,14,15);
}
// B-matrix 32x16 bf16: lane holds column (lane&15); lanes 0-15 K=0..15, 16-31 K=16..31
// -> one contiguous 16-elem load from the (row-major, K-contiguous) weight row.

// ---------------- f32 -> bf16 convert ----------------
__global__ __launch_bounds__(256) void k_cvt_bf16(const float* __restrict__ in,
                                                  bf16_t* __restrict__ out, int n) {
  int i = blockIdx.x * 256 + threadIdx.x;
  if (i < n) out[i] = (bf16_t)in[i];
}

// ---------------- LayerNorm (one wave per 512-row), bf16 out ----------------
__global__ __launch_bounds__(256) void k_layernorm(const float* __restrict__ x,
                                                   const float* __restrict__ g,
                                                   const float* __restrict__ b,
                                                   bf16_t* __restrict__ out, int rows) {
  int row = blockIdx.x * 8 + (threadIdx.x >> 5);
  int lane = threadIdx.x & 31;
  if (row >= rows) return;
  const float* xr = x + (size_t)row * Dm;
  float v[16]; float s = 0.f;
#pragma unroll
  for (int i = 0; i < 16; ++i) { v[i] = xr[lane + i * 32]; s += v[i]; }
#pragma unroll
  for (int o = 16; o > 0; o >>= 1) s += __shfl_xor(s, o, 32);
  float mu = s * (1.f / Dm);
  float var = 0.f;
#pragma unroll
  for (int i = 0; i < 16; ++i) { float d = v[i] - mu; var += d * d; }
#pragma unroll
  for (int o = 16; o > 0; o >>= 1) var += __shfl_xor(var, o, 32);
  float rstd = rsqrtf(var * (1.f / Dm) + 1e-5f);
  bf16_t* orow = out + (size_t)row * Dm;
#pragma unroll
  for (int i = 0; i < 16; ++i) {
    int c = lane + i * 32;
    orow[c] = (bf16_t)((v[i] - mu) * rstd * g[c] + b[c]);
  }
}

// ---------------- WMMA GEMM: C[M,N] = A[M,K] * W[N,K]^T (+bias, act, resid) ---
// block = 4 waves (128 thr); wave -> 32x64 C tile (2 M-subtiles x 4 N-subtiles,
// B fragments reused across both M-subtiles -> ~21 flops/byte from cache).
// grid = (M/128, N/64); all dims exact multiples.
// out row remap: orow = (m / rpb) * obs + (m % rpb)  (identity when rpb >= M)
template <int ACT /*0=none,1=relu,2=sigmoid*/, bool RESID, bool OUTBF>
__global__ __launch_bounds__(128) void k_gemm(const bf16_t* __restrict__ A, int lda,
                                              const bf16_t* __restrict__ W, int ldw,
                                              const float* __restrict__ bias,
                                              const float* __restrict__ resid,
                                              float* __restrict__ outF,
                                              bf16_t* __restrict__ outB, int ldo,
                                              int M, int N, int K, int rpb, int obs) {
  const int wave = threadIdx.x >> 5, lane = threadIdx.x & 31;
  const int lr = lane & 15, lh = lane >> 4;
  const int m0 = blockIdx.x * 128 + wave * 32;
  const int n0 = blockIdx.y * 64;

  v8f acc0[4], acc1[4];
#pragma unroll
  for (int j = 0; j < 4; ++j) { acc0[j] = {}; acc1[j] = {}; }

  const bf16_t* aptr0 = A + (size_t)(m0 + lr) * lda + lh * 16;
  const bf16_t* aptr1 = A + (size_t)(m0 + 16 + lr) * lda + lh * 16;
  const bf16_t* wptr  = W + (size_t)(n0 + lr) * ldw + lh * 16;

  for (int k = 0; k < K; k += 32) {
    v16bf a0 = afrag(aptr0, k);
    v16bf a1 = afrag(aptr1, k);
#pragma unroll
    for (int j = 0; j < 4; ++j) {
      v16bf bfr = *(const v16bf*)(wptr + (size_t)j * 16 * ldw + k);
      acc0[j] = __builtin_amdgcn_wmma_f32_16x16x32_bf16(false, a0, false, bfr,
                                                        (short)0, acc0[j], false, false);
      acc1[j] = __builtin_amdgcn_wmma_f32_16x16x32_bf16(false, a1, false, bfr,
                                                        (short)0, acc1[j], false, false);
    }
  }
#pragma unroll
  for (int sub = 0; sub < 2; ++sub) {
    const int mb = m0 + sub * 16;
#pragma unroll
    for (int j = 0; j < 4; ++j) {
      const int n = n0 + j * 16 + lr;
      const float bv = bias[n];
#pragma unroll
      for (int v = 0; v < 8; ++v) {
        const int m = mb + v + 8 * lh;
        float x = (sub == 0 ? acc0[j][v] : acc1[j][v]) + bv;
        if (RESID) x += resid[(size_t)m * N + n];
        if (ACT == 1) x = x > 0.f ? x : 0.f;
        if (ACT == 2) x = 1.f / (1.f + __expf(-x));
        const size_t orow = (size_t)(m / rpb) * obs + (m % rpb);
        if (OUTBF) outB[orow * ldo + n] = (bf16_t)x;
        else       outF[orow * ldo + n] = x;
      }
    }
  }
}

// ---------------- WMMA attention: one block per (b,h) ----------------
// MODE 0: self (NQ=66, causal on data tokens)   MODE 1: cross (NQ=50)
template <int MODE>
__global__ __launch_bounds__(256) void k_attention(const bf16_t* __restrict__ Q, int q_ld, int q_rpb,
                                                   const bf16_t* __restrict__ KV, int kv_ld,
                                                   int koff, int voff,
                                                   bf16_t* __restrict__ O, int o_ld) {
  constexpr int NQ = (MODE == 0) ? T : CHUNK;
  constexpr int NQT = (NQ + 15) / 16;           // 5 or 4 q-tiles
  const int b = blockIdx.x >> 3, h = blockIdx.x & 7;
  const int tid = threadIdx.x, wave = tid >> 5, lane = tid & 31;
  const int lr = lane & 15, lh = lane >> 4;

  __shared__ alignas(32) bf16_t Vt[64][96];         // V^T, keys padded 66->96
  __shared__ alignas(32) bf16_t Pl[NQT][16][96];    // softmax(P), padded

  // stage V^T (bf16), zero key padding
  {
    const bf16_t* Vb = KV + (size_t)b * T * kv_ld + voff + h * HD;
    const int hd = tid & 63;
    for (int key = tid >> 6; key < 96; key += 4)
      Vt[hd][key] = (key < T) ? Vb[(size_t)key * kv_ld + hd] : (bf16_t)0.f;
  }
  __syncthreads();
  if (wave >= NQT) return;

  const int q0 = wave * 16;
  int qr = q0 + lr; if (qr > NQ - 1) qr = NQ - 1;   // clamp padded q rows
  const bf16_t* qb = Q + ((size_t)b * q_rpb + qr) * q_ld + h * HD + lh * 16;
  const v16bf qa0 = afrag(qb, 0);
  const v16bf qa1 = afrag(qb, 32);

  float sc[5][8];
#pragma unroll
  for (int kt = 0; kt < 5; ++kt) {
    int kr = kt * 16 + lr; if (kr > T - 1) kr = T - 1;
    const bf16_t* kb = KV + ((size_t)b * T + kr) * kv_ld + koff + h * HD + lh * 16;
    v16bf k0 = *(const v16bf*)(kb);
    v16bf k1 = *(const v16bf*)(kb + 32);
    v8f s = {};
    s = __builtin_amdgcn_wmma_f32_16x16x32_bf16(false, qa0, false, k0, (short)0, s, false, false);
    s = __builtin_amdgcn_wmma_f32_16x16x32_bf16(false, qa1, false, k1, (short)0, s, false, false);
#pragma unroll
    for (int v = 0; v < 8; ++v) sc[kt][v] = s[v];
  }

  // mask + softmax per row (row m lives in one 16-lane half -> xor reduce {1,2,4,8})
#pragma unroll
  for (int v = 0; v < 8; ++v) {
    const int q = q0 + v + 8 * lh;
    float mx = -3.0e38f;
#pragma unroll
    for (int kt = 0; kt < 5; ++kt) {
      const int key = kt * 16 + lr;
      float s = sc[kt][v] * 0.125f;                 // 1/sqrt(64)
      bool masked = (MODE == 0) ? (key >= MEMN && q >= MEMN && key > q)
                                : (key >= MEMN && (key - MEMN) > q);
      if (masked)   s = -1.0e9f;
      if (key >= T) s = -3.0e38f;                   // hard-exclude padding
      sc[kt][v] = s;
      mx = fmaxf(mx, s);
    }
#pragma unroll
    for (int o = 8; o > 0; o >>= 1) mx = fmaxf(mx, __shfl_xor(mx, o, 32));
    float sum = 0.f;
#pragma unroll
    for (int kt = 0; kt < 5; ++kt) { float e = __expf(sc[kt][v] - mx); sc[kt][v] = e; sum += e; }
#pragma unroll
    for (int o = 8; o > 0; o >>= 1) sum += __shfl_xor(sum, o, 32);
    const float inv = 1.f / sum;
    const int m = v + 8 * lh;
#pragma unroll
    for (int kt = 0; kt < 5; ++kt) Pl[wave][m][kt * 16 + lr] = (bf16_t)(sc[kt][v] * inv);
    Pl[wave][m][80 + lr] = (bf16_t)0.f;
  }

  // O[16,64] = P[16,96] x V[96,64]
  v8f oc[4];
#pragma unroll
  for (int j = 0; j < 4; ++j) oc[j] = {};
  const bf16_t* pb = &Pl[wave][lr][0] + lh * 16;
#pragma unroll
  for (int ks = 0; ks < 96; ks += 32) {
    v16bf pa = afrag(pb, ks);
#pragma unroll
    for (int j = 0; j < 4; ++j) {
      v16bf vb = *(const v16bf*)(&Vt[j * 16 + lr][ks + lh * 16]);
      oc[j] = __builtin_amdgcn_wmma_f32_16x16x32_bf16(false, pa, false, vb, (short)0, oc[j], false, false);
    }
  }
#pragma unroll
  for (int j = 0; j < 4; ++j)
#pragma unroll
    for (int v = 0; v < 8; ++v) {
      const int q = q0 + v + 8 * lh;
      if (q < NQ)
        O[((size_t)b * q_rpb + q) * o_ld + h * HD + j * 16 + lr] = (bf16_t)oc[j][v];
    }
}

// ---------------- elementwise kernels ----------------
__global__ __launch_bounds__(256) void k_build_combined(float* __restrict__ X,
    const float* __restrict__ mem_state, const float* __restrict__ emb,
    const int* __restrict__ ids, const float* __restrict__ pos, int c) {
  int i = blockIdx.x * 256 + threadIdx.x;
  if (i >= Bb * T * Dm) return;
  int d = i & (Dm - 1), t = (i >> 9) % T, b = i / (T * Dm);
  float v = (t < MEMN) ? mem_state[((size_t)b * MEMN + t) * Dm + d]
                       : emb[(size_t)ids[b * Ll + c * CHUNK + (t - MEMN)] * Dm + d];
  X[i] = v + pos[t * Dm + d];
}

__global__ __launch_bounds__(256) void k_build_ckv(float* __restrict__ CKV,
    const float* __restrict__ proj_mem, const float* __restrict__ enc_out, int c) {
  int i = blockIdx.x * 256 + threadIdx.x;
  if (i >= Bb * T * Dm) return;
  int d = i & (Dm - 1), t = (i >> 9) % T, b = i / (T * Dm);
  CKV[i] = (t < MEMN) ? proj_mem[((size_t)b * MEMN + t) * Dm + d]
                      : enc_out[((size_t)b * Ll + c * CHUNK + (t - MEMN)) * Dm + d];
}

__global__ __launch_bounds__(256) void k_slice_data(float* __restrict__ Dd0,
                                                    const float* __restrict__ X3) {
  int i = blockIdx.x * 256 + threadIdx.x;
  if (i >= Bb * CHUNK * Dm) return;
  int d = i & (Dm - 1), t = (i >> 9) % CHUNK, b = i / (CHUNK * Dm);
  Dd0[i] = X3[((size_t)b * T + MEMN + t) * Dm + d];
}

__global__ __launch_bounds__(256) void k_slice_mem_bf(bf16_t* __restrict__ Mn,
                                                      const float* __restrict__ X3) {
  int i = blockIdx.x * 256 + threadIdx.x;
  if (i >= Bb * MEMN * Dm) return;
  int d = i & (Dm - 1), t = (i >> 9) % MEMN, b = i / (MEMN * Dm);
  Mn[i] = (bf16_t)X3[((size_t)b * T + t) * Dm + d];
}

__global__ __launch_bounds__(256) void k_update_mem(float* __restrict__ Mst,
    const float* __restrict__ X3, const float* __restrict__ G, int c) {
  int i = blockIdx.x * 256 + threadIdx.x;
  if (i >= Bb * MEMN * Dm) return;
  int d = i & (Dm - 1), t = (i >> 9) % MEMN, b = i / (MEMN * Dm);
  float mn = X3[((size_t)b * T + t) * Dm + d];
  if (c == 0) { Mst[i] = mn; }
  else { float g = G[i]; Mst[i] = g * mn + (1.f - g) * Mst[i]; }
}

// ---------------- host-side launch helpers ----------------
static void gemm_b(hipStream_t s, const bf16_t* A, int lda, const bf16_t* W, int ldw,
                   const float* bias, bf16_t* oB, int ldo, int M, int N, int K) {
  k_gemm<0, false, true><<<dim3(M / 128, N / 64), 128, 0, s>>>(A, lda, W, ldw, bias,
      nullptr, nullptr, oB, ldo, M, N, K, M, 0);
}
static void gemm_f(hipStream_t s, const bf16_t* A, int lda, const bf16_t* W, int ldw,
                   const float* bias, float* oF, int ldo, int M, int N, int K) {
  k_gemm<0, false, false><<<dim3(M / 128, N / 64), 128, 0, s>>>(A, lda, W, ldw, bias,
      nullptr, oF, nullptr, ldo, M, N, K, M, 0);
}
static void gemm_relu_b(hipStream_t s, const bf16_t* A, int lda, const bf16_t* W, int ldw,
                        const float* bias, bf16_t* oB, int ldo, int M, int N, int K) {
  k_gemm<1, false, true><<<dim3(M / 128, N / 64), 128, 0, s>>>(A, lda, W, ldw, bias,
      nullptr, nullptr, oB, ldo, M, N, K, M, 0);
}
static void gemm_res_f(hipStream_t s, const bf16_t* A, int lda, const bf16_t* W, int ldw,
                       const float* bias, const float* resid, float* oF, int ldo,
                       int M, int N, int K, int rpb, int obs) {
  k_gemm<0, true, false><<<dim3(M / 128, N / 64), 128, 0, s>>>(A, lda, W, ldw, bias,
      resid, oF, nullptr, ldo, M, N, K, rpb, obs);
}
static void gemm_sig_f(hipStream_t s, const bf16_t* A, int lda, const bf16_t* W, int ldw,
                       const float* bias, float* oF, int ldo, int M, int N, int K) {
  k_gemm<2, false, false><<<dim3(M / 128, N / 64), 128, 0, s>>>(A, lda, W, ldw, bias,
      nullptr, oF, nullptr, ldo, M, N, K, M, 0);
}
static void cvt(hipStream_t s, const float* in, bf16_t* out, int n) {
  k_cvt_bf16<<<(n + 255) / 256, 256, 0, s>>>(in, out, n);
}

extern "C" void kernel_launch(void* const* d_in, const int* in_sizes, int n_in,
                              void* d_out, int out_size, void* d_ws, size_t ws_size,
                              hipStream_t stream) {
  (void)in_sizes; (void)n_in; (void)out_size; (void)ws_size;
  const int*   in_res   = (const int*)  d_in[0];
  const float* enc_out  = (const float*)d_in[1];
  const float* enc_mem  = (const float*)d_in[2];
  const float* emb            = (const float*)d_in[3];
  const float* mem_proj_w     = (const float*)d_in[4];
  const float* mem_proj_b     = (const float*)d_in[5];
  const float* pos_emb        = (const float*)d_in[6];
  const float* sa_in_w        = (const float*)d_in[7];
  const float* sa_in_b        = (const float*)d_in[8];
  const float* sa_out_w       = (const float*)d_in[9];
  const float* sa_out_b       = (const float*)d_in[10];
  const float* rma_ln_attn_g  = (const float*)d_in[11];
  const float* rma_ln_attn_b  = (const float*)d_in[12];
  const float* rma_ln_ffn_g   = (const float*)d_in[13];
  const float* rma_ln_ffn_b   = (const float*)d_in[14];
  const float* rma_ffn_w1     = (const float*)d_in[15];
  const float* rma_ffn_b1     = (const float*)d_in[16];
  const float* rma_ffn_w2     = (const float*)d_in[17];
  const float* rma_ffn_b2     = (const float*)d_in[18];
  const float* enc_mem_proj_w = (const float*)d_in[19];
  const float* enc_mem_proj_b = (const float*)d_in[20];
  const float* ca_in_w        = (const float*)d_in[21];
  const float* ca_in_b        = (const float*)d_in[22];
  const float* ca_out_w       = (const float*)d_in[23];
  const float* ca_out_b       = (const float*)d_in[24];
  const float* ln_cq_g        = (const float*)d_in[25];
  const float* ln_cq_b        = (const float*)d_in[26];
  const float* ln_ckv_g       = (const float*)d_in[27];
  const float* ln_ckv_b       = (const float*)d_in[28];
  const float* ln_ffn_g       = (const float*)d_in[29];
  const float* ln_ffn_b       = (const float*)d_in[30];
  const float* gate_w         = (const float*)d_in[31];
  const float* gate_b         = (const float*)d_in[32];
  const float* ffn_w1         = (const float*)d_in[33];
  const float* ffn_b1         = (const float*)d_in[34];
  const float* ffn_w2         = (const float*)d_in[35];
  const float* ffn_b2         = (const float*)d_in[36];
  float* out = (float*)d_out;

  // ---- deterministic workspace carving ----
  char* p = (char*)d_ws;
  auto alloc = [&](size_t bytes) -> void* {
    void* r = (void*)p; p += (bytes + 255) & ~(size_t)255; return r;
  };
  const int MT = Bb * T;        // 4224
  const int MC = Bb * CHUNK;    // 3200
  const int MM = Bb * MEMN;     // 1024

  bf16_t* w_mem   = (bf16_t*)alloc((size_t)512 * 512 * 2);
  bf16_t* w_emp   = (bf16_t*)alloc((size_t)512 * 512 * 2);
  bf16_t* w_sain  = (bf16_t*)alloc((size_t)1536 * 512 * 2);
  bf16_t* w_saout = (bf16_t*)alloc((size_t)512 * 512 * 2);
  bf16_t* w_rma1  = (bf16_t*)alloc((size_t)2048 * 512 * 2);
  bf16_t* w_rma2  = (bf16_t*)alloc((size_t)512 * 2048 * 2);
  bf16_t* w_cain  = (bf16_t*)alloc((size_t)1536 * 512 * 2);
  bf16_t* w_caout = (bf16_t*)alloc((size_t)512 * 512 * 2);
  bf16_t* w_ffn1  = (bf16_t*)alloc((size_t)2048 * 512 * 2);
  bf16_t* w_ffn2  = (bf16_t*)alloc((size_t)512 * 2048 * 2);
  bf16_t* w_gate  = (bf16_t*)alloc((size_t)512 * 512 * 2);
  bf16_t* encm_bf = (bf16_t*)alloc((size_t)MM * Dm * 2);
  float*  Mst     = (float*) alloc((size_t)MM * Dm * 4);
  float*  Pm      = (float*) alloc((size_t)MM * Dm * 4);
  float*  X       = (float*) alloc((size_t)MT * Dm * 4);
  bf16_t* Xn      = (bf16_t*)alloc((size_t)MT * Dm * 2);
  bf16_t* QKV     = (bf16_t*)alloc((size_t)MT * 1536 * 2);
  bf16_t* Ob      = (bf16_t*)alloc((size_t)MT * Dm * 2);
  float*  X2      = (float*) alloc((size_t)MT * Dm * 4);
  bf16_t* Yn      = (bf16_t*)alloc((size_t)MT * Dm * 2);
  bf16_t* H1      = (bf16_t*)alloc((size_t)MT * 2048 * 2);
  float*  X3      = (float*) alloc((size_t)MT * Dm * 4);
  float*  Dd0     = (float*) alloc((size_t)MC * Dm * 4);
  bf16_t* Mn      = (bf16_t*)alloc((size_t)MM * Dm * 2);
  float*  G       = (float*) alloc((size_t)MM * Dm * 4);
  bf16_t* Qn      = (bf16_t*)alloc((size_t)MC * Dm * 2);
  float*  CKV     = (float*) alloc((size_t)MT * Dm * 4);
  bf16_t* CKVn    = (bf16_t*)alloc((size_t)MT * Dm * 2);
  bf16_t* Qc      = (bf16_t*)alloc((size_t)MC * Dm * 2);
  bf16_t* KVc     = (bf16_t*)alloc((size_t)MT * 1024 * 2);
  bf16_t* O2      = (bf16_t*)alloc((size_t)MC * Dm * 2);
  float*  D2      = (float*) alloc((size_t)MC * Dm * 4);
  bf16_t* D2n     = (bf16_t*)alloc((size_t)MC * Dm * 2);
  bf16_t* H2      = (bf16_t*)alloc((size_t)MC * 2048 * 2);

  // ---- one-time: weights & enc_mem to bf16, initial projections ----
  cvt(stream, mem_proj_w,     w_mem,   512 * 512);
  cvt(stream, enc_mem_proj_w, w_emp,   512 * 512);
  cvt(stream, sa_in_w,        w_sain,  1536 * 512);
  cvt(stream, sa_out_w,       w_saout, 512 * 512);
  cvt(stream, rma_ffn_w1,     w_rma1,  2048 * 512);
  cvt(stream, rma_ffn_w2,     w_rma2,  512 * 2048);
  cvt(stream, ca_in_w,        w_cain,  1536 * 512);
  cvt(stream, ca_out_w,       w_caout, 512 * 512);
  cvt(stream, ffn_w1,         w_ffn1,  2048 * 512);
  cvt(stream, ffn_w2,         w_ffn2,  512 * 2048);
  cvt(stream, gate_w,         w_gate,  512 * 512);
  cvt(stream, enc_mem,        encm_bf, MM * Dm);

  gemm_f(stream, encm_bf, Dm, w_mem, Dm, mem_proj_b,     Mst, Dm, MM, Dm, Dm);
  gemm_f(stream, encm_bf, Dm, w_emp, Dm, enc_mem_proj_b, Pm,  Dm, MM, Dm, Dm);

  const int nT = Bb * T * Dm, nC = Bb * CHUNK * Dm, nM = Bb * MEMN * Dm;

  for (int c = 0; c < NCH; ++c) {
    // --- recurrent-memory self-attention block ---
    k_build_combined<<<(nT + 255) / 256, 256, 0, stream>>>(X, Mst, emb, in_res, pos_emb, c);
    k_layernorm<<<MT / 8, 256, 0, stream>>>(X, rma_ln_attn_g, rma_ln_attn_b, Xn, MT);
    gemm_b(stream, Xn, Dm, w_sain, Dm, sa_in_b, QKV, 1536, MT, 1536, Dm);
    k_attention<0><<<Bb * Hh, 256, 0, stream>>>(QKV, 1536, T, QKV, 1536, 512, 1024, Ob, Dm);
    gemm_res_f(stream, Ob, Dm, w_saout, Dm, sa_out_b, X, X2, Dm, MT, Dm, Dm, MT, 0);
    k_layernorm<<<MT / 8, 256, 0, stream>>>(X2, rma_ln_ffn_g, rma_ln_ffn_b, Yn, MT);
    gemm_relu_b(stream, Yn, Dm, w_rma1, Dm, rma_ffn_b1, H1, 2048, MT, 2048, Dm);
    gemm_res_f(stream, H1, 2048, w_rma2, 2048, rma_ffn_b2, X2, X3, Dm, MT, Dm, 2048, MT, 0);

    // --- gated memory update ---
    k_slice_mem_bf<<<(nM + 255) / 256, 256, 0, stream>>>(Mn, X3);
    gemm_sig_f(stream, Mn, Dm, w_gate, Dm, gate_b, G, Dm, MM, Dm, Dm);
    k_update_mem<<<(nM + 255) / 256, 256, 0, stream>>>(Mst, X3, G, c);

    // --- cross-attention block ---
    k_slice_data<<<(nC + 255) / 256, 256, 0, stream>>>(Dd0, X3);
    k_layernorm<<<MC / 8, 256, 0, stream>>>(Dd0, ln_cq_g, ln_cq_b, Qn, MC);
    k_build_ckv<<<(nT + 255) / 256, 256, 0, stream>>>(CKV, Pm, enc_out, c);
    k_layernorm<<<MT / 8, 256, 0, stream>>>(CKV, ln_ckv_g, ln_ckv_b, CKVn, MT);
    gemm_b(stream, Qn,   Dm, w_cain,             Dm, ca_in_b,       Qc,  Dm,   MC, Dm,   Dm);
    gemm_b(stream, CKVn, Dm, w_cain + 512 * 512, Dm, ca_in_b + 512, KVc, 1024, MT, 1024, Dm);
    k_attention<1><<<Bb * Hh, 256, 0, stream>>>(Qc, Dm, CHUNK, KVc, 1024, 0, 512, O2, Dm);
    gemm_res_f(stream, O2, Dm, w_caout, Dm, ca_out_b, Dd0, D2, Dm, MC, Dm, Dm, MC, 0);

    // --- final FFN, write straight to d_out[b, c*50+t, :] ---
    k_layernorm<<<MC / 8, 256, 0, stream>>>(D2, ln_ffn_g, ln_ffn_b, D2n, MC);
    gemm_relu_b(stream, D2n, Dm, w_ffn1, Dm, ffn_b1, H2, 2048, MC, 2048, Dm);
    gemm_res_f(stream, H2, 2048, w_ffn2, 2048, ffn_b2, D2,
               out + (size_t)c * CHUNK * Dm, Dm, MC, Dm, 2048, CHUNK, Ll);
  }
}
